// MSA_85109071938014
// MI455X (gfx1250) — matrix operand order repaired and checked by reading, not compiled, not run on hardware
//
#include <hip/hip_runtime.h>
#include <math.h>

// ---------------------------------------------------------------------------
// Types for CDNA5 WMMA (wave32): v_wmma_f32_16x16x32_bf16
// ---------------------------------------------------------------------------
typedef __bf16 bf16_t;
typedef __attribute__((ext_vector_type(16))) __bf16 v16bf;
typedef __attribute__((ext_vector_type(8)))  __bf16 v8bf;
typedef __attribute__((ext_vector_type(4)))  __bf16 v4bf;
typedef __attribute__((ext_vector_type(8)))  float  v8f;

// Problem constants (fixed by reference setup_inputs)
#define BATCH   8
#define NWIN    64
#define BNW     512          // B*nW
#define FW      128          // tokens per window
#define ED      512
#define HEADS   8
#define HD      64           // head dim
#define M_TOT   (BNW * FW)   // 65536
#define SCALE_Q 0.125f       // HD^-0.5

// Workspace layout (bytes)
#define XB_OFF      0ull                      // x in bf16            [65536,512]
#define QKVWT_OFF   67108864ull               // qkv_w^T bf16         [1536,512]
#define PROJWT_OFF  68681728ull               // proj_w^T bf16        [512,512]
#define QWS_OFF     69206016ull               // q  bf16 [wh][128][64] (scaled)
#define KWS_OFF     136314880ull              // k  bf16 [wh][128][64]
#define VWS_OFF     203423744ull              // v^T bf16 [wh][64][128]
#define AO_OFF      270532608ull              // attn out bf16 [65536,512]

// ---------------------------------------------------------------------------
// Fragment loaders (layouts per CDNA5 ISA 7.12.2, wave32)
// ---------------------------------------------------------------------------
// A: 16x32 bf16 tile, row-major with leading dim lda.
// lane L: row m = L&15; K-chunks [h*8, h*8+8) and [16+h*8, 16+h*8+8), h=L>>4
__device__ __forceinline__ v16bf load_a_frag(const bf16_t* __restrict__ tile,
                                             int lda, int lane) {
  const int m = lane & 15;
  const int h = lane >> 4;
  const bf16_t* p = tile + m * lda + h * 8;
  v8bf c0 = *(const v8bf*)(p);
  v8bf c1 = *(const v8bf*)(p + 16);
  return __builtin_shufflevector(c0, c1, 0, 1, 2, 3, 4, 5, 6, 7,
                                 8, 9, 10, 11, 12, 13, 14, 15);
}

// B: 32x16 bf16 tile, stored N-major (i.e. B^T row-major, leading dim ldk).
// lane L: col n = L&15; K-chunks [h*8, h*8+8) and [16+h*8, 16+h*8+8), h=L>>4
__device__ __forceinline__ v16bf load_b_frag(const bf16_t* __restrict__ tileT,
                                             int ldk, int lane) {
  const int n = lane & 15;
  const int h = lane >> 4;
  const bf16_t* p = tileT + n * ldk + h * 8;
  v8bf c0 = *(const v8bf*)(p);
  v8bf c1 = *(const v8bf*)(p + 16);
  return __builtin_shufflevector(c0, c1, 0, 1, 2, 3, 4, 5, 6, 7,
                                 8, 9, 10, 11, 12, 13, 14, 15);
}

__device__ __forceinline__ v8f wmma_bf16(v16bf a, v16bf b, v8f c) {
  return __builtin_amdgcn_wmma_f32_16x16x32_bf16(false, a, false, b,
                                                 (short)0, c, false, false);
}

__device__ __forceinline__ v8f zero8() {
  v8f z = {0.f, 0.f, 0.f, 0.f, 0.f, 0.f, 0.f, 0.f};
  return z;
}

// Wave id, forced into an SGPR so downstream tile math is scalar.
__device__ __forceinline__ int wave_id() {
  return __builtin_amdgcn_readfirstlane((int)(threadIdx.x >> 5));
}

// ---------------------------------------------------------------------------
// Prep kernels: fp32 -> bf16 conversion + weight transpose
// ---------------------------------------------------------------------------
__global__ __launch_bounds__(256) void cvt_f32_bf16_kernel(
    const float* __restrict__ in, bf16_t* __restrict__ out, int n4) {
  int i = blockIdx.x * 256 + threadIdx.x;
  if (i < n4) {
    float4 v = ((const float4*)in)[i];
    v4bf o;
    o[0] = (bf16_t)v.x; o[1] = (bf16_t)v.y;
    o[2] = (bf16_t)v.z; o[3] = (bf16_t)v.w;
    ((v4bf*)out)[i] = o;
  }
}

// in: [K,N] f32 row-major -> out: [N,K] bf16 row-major
__global__ __launch_bounds__(256) void transpose_cvt_kernel(
    const float* __restrict__ in, bf16_t* __restrict__ out, int K, int N) {
  int t = blockIdx.x * 256 + threadIdx.x;
  if (t < K * N) {
    int k = t / N, n = t % N;
    out[n * K + k] = (bf16_t)in[t];
  }
}

// ---------------------------------------------------------------------------
// Stage 1: QKV GEMM  [65536,512] x [512,1536] + bias, epilogue scatters
//          q (scaled), k, v^T into head-major workspaces (bf16).
// One wave computes a 64x64 output tile. Grid: 24576 wave-tiles / 8 waves.
// ---------------------------------------------------------------------------
__global__ __launch_bounds__(256) void qkv_gemm_kernel(
    const bf16_t* __restrict__ X,      // [65536,512]
    const bf16_t* __restrict__ WT,     // [1536,512]  (qkv_w^T)
    const float*  __restrict__ bias,   // [1536]
    bf16_t* __restrict__ qws, bf16_t* __restrict__ kws,
    bf16_t* __restrict__ vws) {
  const int lane = threadIdx.x & 31;
  const int gw   = blockIdx.x * 8 + wave_id();   // scalar
  const int TN   = 3 * ED / 64;                  // 24
  const int tn   = gw % TN;                      // scalar
  const int tm   = gw / TN;                      // scalar, < 1024
  const int m0   = tm * 64;
  const int n0   = tn * 64;

  v8f acc[4][4];
#pragma unroll
  for (int i = 0; i < 4; ++i)
#pragma unroll
    for (int j = 0; j < 4; ++j) acc[i][j] = zero8();

  for (int kk = 0; kk < ED; kk += 32) {
    v16bf a[4], b[4];
#pragma unroll
    for (int i = 0; i < 4; ++i)
      a[i] = load_a_frag(X + (size_t)(m0 + i * 16) * ED + kk, ED, lane);
#pragma unroll
    for (int j = 0; j < 4; ++j)
      b[j] = load_b_frag(WT + (size_t)(n0 + j * 16) * ED + kk, ED, lane);
#pragma unroll
    for (int i = 0; i < 4; ++i)
#pragma unroll
      for (int j = 0; j < 4; ++j)
        acc[i][j] = wmma_bf16(a[i], b[j], acc[i][j]);
  }

  // Epilogue. m0 is 64-aligned so the whole 64-row tile sits in one window;
  // each 16-col j-tile sits in one (q3, head) block. With gw scalarized,
  // q3/h/wh and all bases are SGPRs -> scalar branches, tiny lane offsets.
  const int nl = lane & 15, half = lane >> 4;
  const int w  = m0 >> 7;          // scalar: window
  const int fb = m0 & 127;         // scalar: token base (0 or 64)
#pragma unroll
  for (int j = 0; j < 4; ++j) {
    const int ntile = n0 + j * 16;      // scalar
    const int q3 = ntile >> 9;          // scalar: 0:q 1:k 2:v
    const int h  = (ntile >> 6) & 7;    // scalar: head
    const int db = ntile & 63;          // scalar: d base (multiple of 16)
    const int wh = w * HEADS + h;       // scalar
    const int d  = db + nl;             // lane-dependent
    const float bn = bias[ntile + nl];
    if (q3 == 0) {
      bf16_t* base = qws + ((size_t)wh * FW + fb) * HD + d;
#pragma unroll
      for (int i = 0; i < 4; ++i)
#pragma unroll
        for (int r = 0; r < 8; ++r)
          base[(i * 16 + half * 8 + r) * HD] =
              (bf16_t)((acc[i][j][r] + bn) * SCALE_Q);
    } else if (q3 == 1) {
      bf16_t* base = kws + ((size_t)wh * FW + fb) * HD + d;
#pragma unroll
      for (int i = 0; i < 4; ++i)
#pragma unroll
        for (int r = 0; r < 8; ++r)
          base[(i * 16 + half * 8 + r) * HD] = (bf16_t)(acc[i][j][r] + bn);
    } else {
      bf16_t* base = vws + ((size_t)wh * HD + d) * FW + fb;  // V transposed
#pragma unroll
      for (int i = 0; i < 4; ++i)
#pragma unroll
        for (int r = 0; r < 8; ++r)
          base[i * 16 + half * 8 + r] = (bf16_t)(acc[i][j][r] + bn);
    }
  }
}

// ---------------------------------------------------------------------------
// Stage 2: attention per (window, head). 128 threads = 4 waves.
// Phase 1: S = Q K^T (WMMA) -> LDS (f32, stride 132)
// Phase 2: row softmax with additive mask -> P (bf16) in LDS
// Phase 3: O = P V^T (WMMA, A-frags from LDS) -> ao (bf16)
// ---------------------------------------------------------------------------
#define S_STRIDE 132
__global__ __launch_bounds__(128) void attn_kernel(
    const bf16_t* __restrict__ qws, const bf16_t* __restrict__ kws,
    const bf16_t* __restrict__ vws, const float* __restrict__ mask,
    bf16_t* __restrict__ ao) {
  extern __shared__ char smem[];
  float*  S = (float*)smem;                                    // [128][132]
  bf16_t* P = (bf16_t*)(smem + FW * S_STRIDE * sizeof(float)); // [128][128]

  const int wh   = blockIdx.x;          // 0..4095
  const int w    = wh >> 3;
  const int h    = wh & 7;
  const int b    = w >> 6;              // w / nW
  const int wv   = wave_id();           // scalar
  const int lane = threadIdx.x & 31;

  const bf16_t* Q  = qws + (size_t)wh * FW * HD;
  const bf16_t* K  = kws + (size_t)wh * FW * HD;
  const bf16_t* VT = vws + (size_t)wh * HD * FW;
  const float*  Mk = mask + ((size_t)(b * HEADS + h)) * FW * FW;

  // ---- Phase 1: S = Q K^T, wave wv owns rows [wv*32, wv*32+32) ----
  {
    v8f acc[2][8];
#pragma unroll
    for (int i = 0; i < 2; ++i)
#pragma unroll
      for (int j = 0; j < 8; ++j) acc[i][j] = zero8();
#pragma unroll
    for (int kk = 0; kk < HD; kk += 32) {
      v16bf a[2], bfr[8];
#pragma unroll
      for (int i = 0; i < 2; ++i)
        a[i] = load_a_frag(Q + (size_t)(wv * 32 + i * 16) * HD + kk, HD, lane);
#pragma unroll
      for (int j = 0; j < 8; ++j)
        bfr[j] = load_b_frag(K + (size_t)(j * 16) * HD + kk, HD, lane);
#pragma unroll
      for (int i = 0; i < 2; ++i)
#pragma unroll
        for (int j = 0; j < 8; ++j)
          acc[i][j] = wmma_bf16(a[i], bfr[j], acc[i][j]);
    }
    const int nl = lane & 15, half = lane >> 4;
    float* srow = S + (wv * 32 + half * 8) * S_STRIDE + nl;
#pragma unroll
    for (int i = 0; i < 2; ++i)
#pragma unroll
      for (int j = 0; j < 8; ++j)
#pragma unroll
        for (int r = 0; r < 8; ++r)
          srow[(i * 16 + r) * S_STRIDE + j * 16] = acc[i][j][r];
  }
  __syncthreads();

  // ---- Phase 2: per-row softmax (one row per thread) ----
  {
    const int q = threadIdx.x;          // 0..127
    float* srow = S + q * S_STRIDE;
    const float* mrow = Mk + (size_t)q * FW;
    float mx = -3.4e38f;
    for (int k = 0; k < FW; ++k) {
      float v = srow[k] + mrow[k];
      srow[k] = v;
      mx = fmaxf(mx, v);
    }
    float sum = 0.f;
    for (int k = 0; k < FW; ++k) {
      float e = __expf(srow[k] - mx);
      srow[k] = e;
      sum += e;
    }
    const float inv = 1.f / sum;
    for (int k = 0; k < FW; ++k) P[q * FW + k] = (bf16_t)(srow[k] * inv);
  }
  __syncthreads();

  // ---- Phase 3: O = P V^T, wave wv owns rows [wv*32, +32), cols 64 ----
  {
    v8f acc[2][4];
#pragma unroll
    for (int i = 0; i < 2; ++i)
#pragma unroll
      for (int j = 0; j < 4; ++j) acc[i][j] = zero8();
#pragma unroll
    for (int kk = 0; kk < FW; kk += 32) {
      v16bf a[2], bfr[4];
#pragma unroll
      for (int i = 0; i < 2; ++i)
        a[i] = load_a_frag(P + (size_t)(wv * 32 + i * 16) * FW + kk, FW, lane);
#pragma unroll
      for (int j = 0; j < 4; ++j)
        bfr[j] = load_b_frag(VT + (size_t)(j * 16) * FW + kk, FW, lane);
#pragma unroll
      for (int i = 0; i < 2; ++i)
#pragma unroll
        for (int j = 0; j < 4; ++j)
          acc[i][j] = wmma_bf16(a[i], bfr[j], acc[i][j]);
    }
    const int nl = lane & 15, half = lane >> 4;
    // ao[(w*FW + f)*ED + h*HD + d], f = wv*32 + i*16 + half*8 + r, d = j*16+nl
    bf16_t* base =
        ao + ((size_t)(w * FW + wv * 32 + half * 8)) * ED + h * HD + nl;
#pragma unroll
    for (int i = 0; i < 2; ++i)
#pragma unroll
      for (int j = 0; j < 4; ++j)
#pragma unroll
        for (int r = 0; r < 8; ++r)
          base[(size_t)(i * 16 + r) * ED + j * 16] = (bf16_t)acc[i][j][r];
  }
}

// ---------------------------------------------------------------------------
// Stage 3: output projection  [65536,512] x [512,512] + bias -> f32 out
// ---------------------------------------------------------------------------
__global__ __launch_bounds__(256) void proj_gemm_kernel(
    const bf16_t* __restrict__ A,     // attn out bf16 [65536,512]
    const bf16_t* __restrict__ WT,    // proj_w^T bf16 [512,512]
    const float*  __restrict__ bias,  // [512]
    float* __restrict__ out) {
  const int lane = threadIdx.x & 31;
  const int gw   = blockIdx.x * 8 + wave_id();   // scalar
  const int TN   = ED / 64;                      // 8
  const int tn   = gw % TN;
  const int tm   = gw / TN;                      // < 1024
  const int m0   = tm * 64;
  const int n0   = tn * 64;

  v8f acc[4][4];
#pragma unroll
  for (int i = 0; i < 4; ++i)
#pragma unroll
    for (int j = 0; j < 4; ++j) acc[i][j] = zero8();

  for (int kk = 0; kk < ED; kk += 32) {
    v16bf a[4], b[4];
#pragma unroll
    for (int i = 0; i < 4; ++i)
      a[i] = load_a_frag(A + (size_t)(m0 + i * 16) * ED + kk, ED, lane);
#pragma unroll
    for (int j = 0; j < 4; ++j)
      b[j] = load_b_frag(WT + (size_t)(n0 + j * 16) * ED + kk, ED, lane);
#pragma unroll
    for (int i = 0; i < 4; ++i)
#pragma unroll
      for (int j = 0; j < 4; ++j)
        acc[i][j] = wmma_bf16(a[i], b[j], acc[i][j]);
  }

  const int nl = lane & 15, half = lane >> 4;
  float* base = out + (size_t)(m0 + half * 8) * ED + n0 + nl;
  const float bn = bias[n0 + nl];         // j-offset folded below
#pragma unroll
  for (int j = 0; j < 4; ++j) {
    const float bj = bias[n0 + j * 16 + nl];
#pragma unroll
    for (int i = 0; i < 4; ++i)
#pragma unroll
      for (int r = 0; r < 8; ++r)
        base[(size_t)(i * 16 + r) * ED + j * 16] = acc[i][j][r] + bj;
  }
  (void)bn;
}

// ---------------------------------------------------------------------------
// Launch
// ---------------------------------------------------------------------------
extern "C" void kernel_launch(void* const* d_in, const int* in_sizes, int n_in,
                              void* d_out, int out_size, void* d_ws,
                              size_t ws_size, hipStream_t stream) {
  const float* x      = (const float*)d_in[0];
  const float* adj    = (const float*)d_in[1];
  const float* qkv_w  = (const float*)d_in[2];
  const float* qkv_b  = (const float*)d_in[3];
  const float* proj_w = (const float*)d_in[4];
  const float* proj_b = (const float*)d_in[5];
  float* out = (float*)d_out;

  char* ws = (char*)d_ws;
  bf16_t* xb     = (bf16_t*)(ws + XB_OFF);
  bf16_t* qkvwT  = (bf16_t*)(ws + QKVWT_OFF);
  bf16_t* projwT = (bf16_t*)(ws + PROJWT_OFF);
  bf16_t* qws    = (bf16_t*)(ws + QWS_OFF);
  bf16_t* kws    = (bf16_t*)(ws + KWS_OFF);
  bf16_t* vws    = (bf16_t*)(ws + VWS_OFF);
  bf16_t* ao     = (bf16_t*)(ws + AO_OFF);

  // Prep: x -> bf16 (33.5M elems as float4), weight transposes
  cvt_f32_bf16_kernel<<<(M_TOT * ED / 4 + 255) / 256, 256, 0, stream>>>(
      x, xb, M_TOT * ED / 4);
  transpose_cvt_kernel<<<(ED * 3 * ED + 255) / 256, 256, 0, stream>>>(
      qkv_w, qkvwT, ED, 3 * ED);
  transpose_cvt_kernel<<<(ED * ED + 255) / 256, 256, 0, stream>>>(
      proj_w, projwT, ED, ED);

  // Stage 1: QKV projection (24576 wave-tiles, 8 waves per block)
  qkv_gemm_kernel<<<(M_TOT / 64) * (3 * ED / 64) / 8, 256, 0, stream>>>(
      xb, qkvwT, qkv_b, qws, kws, vws);

  // Stage 2: attention, one block per (window, head)
  const size_t attn_lds = FW * S_STRIDE * sizeof(float) + FW * FW * sizeof(bf16_t);
  attn_kernel<<<BNW * HEADS, 128, attn_lds, stream>>>(qws, kws, vws, adj, ao);

  // Stage 3: output projection (8192 wave-tiles)
  proj_gemm_kernel<<<(M_TOT / 64) * (ED / 64) / 8, 256, 0, stream>>>(
      ao, projwT, proj_b, out);
}
